// RadianceField_58411555226352
// MI455X (gfx1250) — matrix operand order repaired and checked by reading, not compiled, not run on hardware
//
#include <hip/hip_runtime.h>
#include <hip/hip_bf16.h>
#include <math.h>

typedef __attribute__((ext_vector_type(16))) _Float16 v16h;
typedef __attribute__((ext_vector_type(8)))  _Float16 v8h;
typedef __attribute__((ext_vector_type(8)))  float    v8f;

#define BATCH    262144
#define BM       128      // samples per block
#define NTHREADS 512      // 16 wave32
#define NWAVES   16
#define SA       264      // bufA row stride in halfs (padded vs 256)
#define SB       328      // bufB row stride in halfs (padded vs 320)
#define BIAS_FLOATS 4096  // ws: first 4096 floats = biases, then f16 weights

// f16 weight offsets in workspace (halfs), layers padded to K%32==0
enum : int {
  W_L0 = 0,       W_L1 = 16384,  W_L2 = 81920,  W_L3 = 147456, W_L4 = 212992,
  W_L5 = 278528,  W_L6 = 360448, W_L7 = 425984, W_SG = 491520, W_L8 = 491776,
  W_L9 = 557312,  W_L10= 594176, W_L11= 610560, W_L12= 626944, W_RGB= 643328
};
// f32 bias offsets in workspace (floats)
enum : int {
  B_L0=0,B_L1=256,B_L2=512,B_L3=768,B_L4=1024,B_L5=1280,B_L6=1536,B_L7=1792,
  B_SG=2048,B_L8=2049,B_L9=2305,B_L10=2433,B_L11=2561,B_L12=2689,B_RGB=2817
};

// ---- pass 1: cast f32 weights -> padded f16 (optionally WMMA-fragment-swizzled) ----
// Swizzled layout: element (n,k) -> tile (n>>4, k>>5); within tile stored as
// [lane][16 halfs] with lane = (n&15) + (k&16), half = k&15  => a wave's
// B-fragment is one contiguous, fully-coalesced 1KB block.
__global__ void convert_layer_kernel(const float* __restrict__ W, const float* __restrict__ b,
                                     _Float16* __restrict__ dW, float* __restrict__ db,
                                     int dout, int din, int kp, int swizzle) {
  const int stride = gridDim.x * blockDim.x;
  const int t0 = blockIdx.x * blockDim.x + threadIdx.x;
  const int total = dout * kp;
  const int kblocks = kp >> 5;
  for (int i = t0; i < total; i += stride) {
    const int n = i / kp, k = i - n * kp;
    const _Float16 v = (k < din) ? (_Float16)W[n * din + k] : (_Float16)0.0f;
    int dsti;
    if (swizzle) {
      const int ntile = n >> 4, nl = n & 15, kb = k >> 5, kl = k & 31;
      const int lane = nl + (kl & 16);
      dsti = ((ntile * kblocks + kb) * 32 + lane) * 16 + (kl & 15);
    } else {
      dsti = i;
    }
    dW[dsti] = v;
  }
  for (int i = t0; i < dout; i += stride) db[i] = b[i];
}

// NeRF frequency encoding element c of [sin(v*f_l), cos(v*f_l)] blocks of 6
__device__ __forceinline__ float enc_component(float v0, float v1, float v2, int c) {
  const int l = c / 6, r = c - l * 6;
  const float f = 3.14159265358979323846f * (float)(1 << l);
  const float v = (r == 0 || r == 3) ? v0 : ((r == 1 || r == 4) ? v1 : v2);
  return (r < 3) ? sinf(v * f) : cosf(v * f);
}

// One dense layer: dst[BM][N] = act( src[BM][K] @ W^T + b ) via WMMA f16->f32.
// 16 waves; N=256 -> 1 col tile/wave, 8 row tiles; N=128 -> 2 waves share a col
// tile, 4 row tiles each. Weights are fragment-swizzled f16 in global (L2-hot).
template<int K, int N, int SS, int DS, bool RELU>
__device__ __forceinline__ void mlp_layer(const _Float16* __restrict__ src,
                                          _Float16* __restrict__ dst,
                                          const _Float16* __restrict__ W,
                                          const float* __restrict__ bias,
                                          int tid) {
  constexpr int KB  = K / 32;              // k-blocks
  constexpr int WPC = NWAVES * 16 / N;     // waves per column tile (1 or 2)
  constexpr int NR  = 8 / WPC;             // row tiles per wave
  const int wave = tid >> 5;
  const int lane = tid & 31;
  const int l16  = lane & 15;
  const int hi   = lane >> 4;              // 0 or 1 (half-wave select)
  const int ct   = wave / WPC;             // column tile
  const int r0   = (wave % WPC) * NR;      // first row tile

  v8f zero = {0.f,0.f,0.f,0.f,0.f,0.f,0.f,0.f};
  v8f acc[NR];
#pragma unroll
  for (int r = 0; r < NR; ++r) acc[r] = zero;

  const _Float16* wp = W + ((size_t)ct * KB) * 512 + lane * 16;  // 512 halfs per 16x32 tile
  for (int kb = 0; kb < KB; ++kb) {
    const v16h bfrag = *(const v16h*)(wp + (size_t)kb * 512);
    const int k0 = kb * 32;
#pragma unroll
    for (int r = 0; r < NR; ++r) {
      // A fragment: row m = l16 within tile; halfs K=k0+(hi?16:0).. 16 contiguous
      const int m = (r0 + r) * 16 + l16;
      const _Float16* sp = src + m * SS + k0 + hi * 16;
      v8h alo = *(const v8h*)sp;
      v8h ahi = *(const v8h*)(sp + 8);
      v16h a = __builtin_shufflevector(alo, ahi, 0,1,2,3,4,5,6,7,8,9,10,11,12,13,14,15);
      acc[r] = __builtin_amdgcn_wmma_f32_16x16x32_f16(
          false, a, false, bfrag, (short)0, acc[r], false, false);
    }
  }
  // D layout: col = l16, row = v + (hi?8:0) within tile; bias + optional ReLU
  const int n = ct * 16 + l16;
  const float bv = bias[n];
#pragma unroll
  for (int r = 0; r < NR; ++r) {
#pragma unroll
    for (int v = 0; v < 8; ++v) {
      const int m = (r0 + r) * 16 + v + hi * 8;
      float val = acc[r][v] + bv;
      if (RELU) val = fmaxf(val, 0.0f);
      dst[m * DS + n] = (_Float16)val;
    }
  }
}

__global__ __launch_bounds__(NTHREADS)
void nerf_forward_kernel(const float* __restrict__ x, const float* __restrict__ d,
                         const _Float16* __restrict__ wh, const float* __restrict__ wf,
                         float* __restrict__ out) {
  extern __shared__ _Float16 smem[];
  _Float16* bufB = smem;            // [BM][SB]
  _Float16* bufA = smem + BM * SB;  // [BM][SA]
  const int tid = threadIdx.x;
  const int g0  = blockIdx.x * BM;

  // e_x -> bufA[:, 0:64) (cols 60..63 zero-padded)
  for (int idx = tid; idx < BM * 64; idx += NTHREADS) {
    const int s = idx >> 6, c = idx & 63;
    const float* xv = x + (size_t)(g0 + s) * 3;
    const float val = (c < 60) ? enc_component(xv[0], xv[1], xv[2], c) : 0.0f;
    bufA[s * SA + c] = (_Float16)val;
  }
  __syncthreads();
  mlp_layer< 64,256,SA,SB,true >(bufA, bufB, wh+W_L0,  wf+B_L0,  tid); __syncthreads();
  mlp_layer<256,256,SB,SA,true >(bufB, bufA, wh+W_L1,  wf+B_L1,  tid); __syncthreads();
  mlp_layer<256,256,SA,SB,true >(bufA, bufB, wh+W_L2,  wf+B_L2,  tid); __syncthreads();
  mlp_layer<256,256,SB,SA,true >(bufB, bufA, wh+W_L3,  wf+B_L3,  tid); __syncthreads();
  mlp_layer<256,256,SA,SB,true >(bufA, bufB, wh+W_L4,  wf+B_L4,  tid);
  // skip concat: e_x -> bufB[:, 256:320)
  for (int idx = tid; idx < BM * 64; idx += NTHREADS) {
    const int s = idx >> 6, c = idx & 63;
    const float* xv = x + (size_t)(g0 + s) * 3;
    const float val = (c < 60) ? enc_component(xv[0], xv[1], xv[2], c) : 0.0f;
    bufB[s * SB + 256 + c] = (_Float16)val;
  }
  __syncthreads();
  mlp_layer<320,256,SB,SA,true >(bufB, bufA, wh+W_L5,  wf+B_L5,  tid); __syncthreads();
  mlp_layer<256,256,SA,SB,true >(bufA, bufB, wh+W_L6,  wf+B_L6,  tid); __syncthreads();
  mlp_layer<256,256,SB,SA,true >(bufB, bufA, wh+W_L7,  wf+B_L7,  tid); __syncthreads();
  // sigma head: relu(h7 . w + b)
  if (tid < BM) {
    const _Float16* h = bufA + tid * SA;
    const _Float16* w = wh + W_SG;
    float a = wf[B_SG];
    for (int k = 0; k < 256; ++k) a += (float)h[k] * (float)w[k];
    out[3 * BATCH + g0 + tid] = fmaxf(a, 0.0f);
  }
  mlp_layer<256,256,SA,SB,false>(bufA, bufB, wh+W_L8,  wf+B_L8,  tid);  // no activation
  // view concat: e_d -> bufB[:, 256:320) (cols 24.. zero)
  for (int idx = tid; idx < BM * 64; idx += NTHREADS) {
    const int s = idx >> 6, c = idx & 63;
    const float* dv = d + (size_t)(g0 + s) * 3;
    const float val = (c < 24) ? enc_component(dv[0], dv[1], dv[2], c) : 0.0f;
    bufB[s * SB + 256 + c] = (_Float16)val;
  }
  __syncthreads();
  mlp_layer<288,128,SB,SA,true >(bufB, bufA, wh+W_L9,  wf+B_L9,  tid); __syncthreads();
  mlp_layer<128,128,SA,SB,true >(bufA, bufB, wh+W_L10, wf+B_L10, tid); __syncthreads();
  mlp_layer<128,128,SB,SA,true >(bufB, bufA, wh+W_L11, wf+B_L11, tid); __syncthreads();
  mlp_layer<128,128,SA,SB,true >(bufA, bufB, wh+W_L12, wf+B_L12, tid); __syncthreads();
  // rgb head: sigmoid(h12 . W_rgb^T + b)
  for (int idx = tid; idx < BM * 3; idx += NTHREADS) {
    const int s = idx & 127, ch = idx >> 7;
    const _Float16* h = bufB + s * SB;
    const _Float16* w = wh + W_RGB + ch * 128;
    float a = wf[B_RGB + ch];
    for (int k = 0; k < 128; ++k) a += (float)h[k] * (float)w[k];
    out[(size_t)(g0 + s) * 3 + ch] = 1.0f / (1.0f + expf(-a));
  }
}

extern "C" void kernel_launch(void* const* d_in, const int* in_sizes, int n_in,
                              void* d_out, int out_size, void* d_ws, size_t ws_size,
                              hipStream_t stream) {
  const float* x   = (const float*)d_in[0];
  const float* dir = (const float*)d_in[1];
  float*     wsf = (float*)d_ws;                   // biases
  _Float16*  wsh = (_Float16*)(wsf + BIAS_FLOATS); // f16 weights

  struct LD { int dout, din, kp, woff, boff, swz; };
  static const LD L[15] = {
    {256, 60, 64,W_L0 ,B_L0 ,1},{256,256,256,W_L1 ,B_L1 ,1},{256,256,256,W_L2 ,B_L2 ,1},
    {256,256,256,W_L3 ,B_L3 ,1},{256,256,256,W_L4 ,B_L4 ,1},{256,316,320,W_L5 ,B_L5 ,1},
    {256,256,256,W_L6 ,B_L6 ,1},{256,256,256,W_L7 ,B_L7 ,1},{  1,256,256,W_SG ,B_SG ,0},
    {256,256,256,W_L8 ,B_L8 ,1},{128,280,288,W_L9 ,B_L9 ,1},{128,128,128,W_L10,B_L10,1},
    {128,128,128,W_L11,B_L11,1},{128,128,128,W_L12,B_L12,1},{  3,128,128,W_RGB,B_RGB,0}
  };
  for (int i = 0; i < 15; ++i) {
    const float* W = (const float*)d_in[2 + 2*i];
    const float* b = (const float*)d_in[3 + 2*i];
    int total = L[i].dout * L[i].kp;
    int grid = (total + 255) / 256; if (grid > 512) grid = 512;
    convert_layer_kernel<<<grid, 256, 0, stream>>>(W, b, wsh + L[i].woff, wsf + L[i].boff,
                                                   L[i].dout, L[i].din, L[i].kp, L[i].swz);
  }
  const size_t lds_bytes = (size_t)(BM * SB + BM * SA) * sizeof(_Float16); // 151552
  nerf_forward_kernel<<<BATCH / BM, NTHREADS, lds_bytes, stream>>>(
      x, dir, wsh, wsf, (float*)d_out);
}
